// MultiHeadAttention_63977832841803
// MI455X (gfx1250) — matrix operand order repaired and checked by reading, not compiled, not run on hardware
//
#include <hip/hip_runtime.h>
#include <math.h>

// ---------------------------------------------------------------------------
// Problem constants (from reference)
// ---------------------------------------------------------------------------
#define HH   8
#define BB   16
#define GG   512
#define NQ   512
#define DD   512
#define KD   64
#define VDIM 64
#define EE   512
#define CTOT 2048            // 4*G concatenated score columns
#define MROWS (BB*NQ)        // 8192 rows for projection / output GEMMs
#define QKSZ ((size_t)HH*MROWS*KD)   // elements per branch Q or K buffer
#define NORMF 0.125f         // 1/sqrt(64)

typedef __attribute__((ext_vector_type(16))) __bf16 v16bf;
typedef __attribute__((ext_vector_type(8)))  __bf16 v8bf;
typedef __attribute__((ext_vector_type(8)))  float  v8f;

struct W12 { const float* p[12]; };

// ---------------------------------------------------------------------------
// Helpers
// ---------------------------------------------------------------------------
__device__ inline v8f zero8() {
  v8f z;
#pragma unroll
  for (int i = 0; i < 8; ++i) z[i] = 0.0f;
  return z;
}

__device__ inline v8f wmma_bf16(v16bf a, v16bf b, v8f c) {
  // D = A(16x32 bf16) * B(32x16 bf16) + C(16x16 f32)
  return __builtin_amdgcn_wmma_f32_16x16x32_bf16(
      false, a, false, b, (short)0, c, false, false);
}

// A-fragment (16x32 bf16) load, contiguous k order (B operands use the same):
// lane<16 : row = lane,    K elements {kb..kb+15}
// lane>=16: row = lane-16, K elements {kb+16..kb+31}
// rowptr already points at this lane's row; kbase is the 32-wide K chunk base.
__device__ inline v16bf load_a_frag(const __bf16* rowptr, int half, int kbase) {
  const __bf16* p = rowptr + kbase + half * 16;
  v8bf lo = *(const v8bf*)(p);
  v8bf hi = *(const v8bf*)(p + 8);
  v16bf a;
#pragma unroll
  for (int i = 0; i < 8; ++i) { a[i] = lo[i]; a[i + 8] = hi[i]; }
  return a;
}

__device__ inline float rmax16(float x) {
  x = fmaxf(x, __shfl_xor(x, 1, 32));
  x = fmaxf(x, __shfl_xor(x, 2, 32));
  x = fmaxf(x, __shfl_xor(x, 4, 32));
  x = fmaxf(x, __shfl_xor(x, 8, 32));
  return x;
}
__device__ inline float rsum16(float x) {
  x += __shfl_xor(x, 1, 32);
  x += __shfl_xor(x, 2, 32);
  x += __shfl_xor(x, 4, 32);
  x += __shfl_xor(x, 8, 32);
  return x;
}

// ---------------------------------------------------------------------------
// 1) f32 -> bf16 conversion of q
// ---------------------------------------------------------------------------
__global__ void qconv_kernel(const float* __restrict__ q, __bf16* __restrict__ qb) {
  int idx = blockIdx.x * blockDim.x + threadIdx.x;
  if (idx < MROWS * DD) qb[idx] = (__bf16)q[idx];
}

// 2) Projection weights: transpose (H,D,KD) -> Wt[m][(h*64+k)][d], bf16
__global__ void wconv_kernel(W12 w, __bf16* __restrict__ Wt) {
  int idx = blockIdx.x * blockDim.x + threadIdx.x;   // 12*8*64*512
  if (idx >= 12 * 8 * 64 * 512) return;
  int d = idx & 511;
  int k = (idx >> 9) & 63;
  int h = (idx >> 15) & 7;
  int m = idx >> 18;
  Wt[idx] = (__bf16)w.p[m][((size_t)h * DD + d) * KD + k];
}

// 3) W_out (H,KD,E) -> WoT[e][(h*64+v)], bf16
__global__ void woutconv_kernel(const float* __restrict__ wo, __bf16* __restrict__ WoT) {
  int idx = blockIdx.x * blockDim.x + threadIdx.x;   // 512*512
  if (idx >= EE * HH * KD) return;
  int c = idx & 511;          // h*64+v
  int e = idx >> 9;
  WoT[idx] = (__bf16)wo[(size_t)c * EE + e];
}

// 4) Pack 4 boolean masks into bits: mb[b][q][w], word w covers cols w*32..+31
//    column c = r*512+g:  r0=sparse, r1=att^T, r2=att(raw view), r3=group
__global__ void maskpack_kernel(const int* __restrict__ att,
                                const int* __restrict__ grp,
                                const int* __restrict__ sp,
                                unsigned* __restrict__ mb) {
  int idx = blockIdx.x * blockDim.x + threadIdx.x;   // 16*512*64
  if (idx >= BB * NQ * (CTOT / 32)) return;
  int w = idx & 63;
  int q = (idx >> 6) & 511;
  int b = idx >> 15;
  unsigned word = 0;
#pragma unroll 4
  for (int j = 0; j < 32; ++j) {
    int c = w * 32 + j;
    int r = c >> 9, g = c & 511;
    int v;
    if (r == 0)      v = sp[((size_t)b * NQ + q) * GG + g];
    else if (r == 1) v = att[((size_t)b * GG + g) * NQ + q];
    else if (r == 2) v = att[((size_t)b * NQ + q) * GG + g];
    else             v = grp[((size_t)b * NQ + q) * GG + g];
    word |= ((unsigned)(v != 0)) << j;
  }
  mb[idx] = word;
}

// ---------------------------------------------------------------------------
// 5) Projections: 12 GEMMs (8192x512)x(512x64). One wave = one 16x16 C tile.
//    qk=0 -> Q[r], qk=1 -> K[r], qk=2 -> V written transposed into Vt[h][b][v][c]
// ---------------------------------------------------------------------------
__global__ __launch_bounds__(256) void proj_kernel(
    const __bf16* __restrict__ qb, const __bf16* __restrict__ Wt,
    __bf16* __restrict__ Qb, __bf16* __restrict__ Kb, __bf16* __restrict__ Vt) {
  int lane = threadIdx.x & 31;
  int half = lane >> 4, lan = lane & 15;
  int w = blockIdx.x * 8 + (threadIdx.x >> 5);
  int ntile = w & 3;   w >>= 2;
  int mtile = w & 511; w >>= 9;
  int h = w & 7;       w >>= 3;
  int qk = w % 3, r = w / 3;
  int mrow = mtile * 16;

  const __bf16* arow = qb + (size_t)(mrow + lan) * DD;
  const __bf16* brow = Wt + ((size_t)((r * 3 + qk) * 8 + h) * 64 + ntile * 16 + lan) * DD;

  v8f c = zero8();
#pragma unroll 4
  for (int kk = 0; kk < DD / 32; ++kk) {
    v16bf a = load_a_frag(arow, half, kk * 32);
    v16bf b = *(const v16bf*)(brow + kk * 32 + half * 16);
    c = wmma_bf16(a, b, c);
  }

  if (qk == 2) {
    int bb = mrow >> 9, gg = mrow & 511;
#pragma unroll
    for (int j = 0; j < 8; ++j) {
      int g = gg + j + 8 * half;
      int v = ntile * 16 + lan;
      Vt[((size_t)(h * BB + bb) * VDIM + v) * CTOT + r * GG + g] = (__bf16)c[j];
    }
  } else {
    __bf16* dst = (qk == 0 ? Qb : Kb) + (size_t)r * QKSZ;
#pragma unroll
    for (int j = 0; j < 8; ++j) {
      int row = mrow + j + 8 * half;
      dst[((size_t)h * MROWS + row) * KD + ntile * 16 + lan] = (__bf16)c[j];
    }
  }
}

// ---------------------------------------------------------------------------
// 6) Fused flash attention: one wave = (h, b, 16-row q-tile).
//    Streams the 2048 concatenated masked columns with online softmax,
//    P re-staged through per-wave LDS into A-fragment layout, P*V via WMMA.
// ---------------------------------------------------------------------------
__global__ __launch_bounds__(128) void attn_kernel(
    const __bf16* __restrict__ Qb, const __bf16* __restrict__ Kb,
    const __bf16* __restrict__ Vt, const unsigned* __restrict__ mb,
    __bf16* __restrict__ heads) {
  __shared__ __align__(32) __bf16 plds[4][16 * 32];
  int wave = threadIdx.x >> 5;
  int lane = threadIdx.x & 31;
  int half = lane >> 4, lan = lane & 15;
  int wid = blockIdx.x * 4 + wave;
  int qtile = wid & 31; wid >>= 5;
  int b = wid & 15;     wid >>= 4;
  int h = wid;                         // 0..7
  int q0 = qtile * 16;

  __bf16* myp = &plds[wave][0];

  float m[8], l[8];
  v8f O[4];
#pragma unroll
  for (int j = 0; j < 8; ++j) { m[j] = -__builtin_inff(); l[j] = 0.0f; }
#pragma unroll
  for (int vt = 0; vt < 4; ++vt) O[vt] = zero8();

  const __bf16* Vb_base = Vt + (size_t)(h * BB + b) * VDIM * CTOT;

  for (int r = 0; r < 4; ++r) {
    const __bf16* qrow = Qb + (size_t)r * QKSZ +
                         ((size_t)h * MROWS + b * NQ + q0 + lan) * KD;
    v16bf qa0 = load_a_frag(qrow, half, 0);
    v16bf qa1 = load_a_frag(qrow, half, 32);
    const __bf16* Kmat = Kb + (size_t)r * QKSZ + ((size_t)h * MROWS + (size_t)b * NQ) * KD;

    for (int gt = 0; gt < GG / 32; ++gt) {
      int cbase = r * GG + gt * 32;
      // --- scores: two 16x16 tiles (cols cbase..+15 and +16..+31)
      v8f s0 = zero8(), s1 = zero8();
      {
        const __bf16* k0 = Kmat + (size_t)(gt * 32 + lan) * KD + half * 16;
        v16bf kb00 = *(const v16bf*)(k0);
        v16bf kb01 = *(const v16bf*)(k0 + 32);
        s0 = wmma_bf16(qa0, kb00, s0);
        s0 = wmma_bf16(qa1, kb01, s0);
        const __bf16* k1 = Kmat + (size_t)(gt * 32 + 16 + lan) * KD + half * 16;
        v16bf kb10 = *(const v16bf*)(k1);
        v16bf kb11 = *(const v16bf*)(k1 + 32);
        s1 = wmma_bf16(qa0, kb10, s1);
        s1 = wmma_bf16(qa1, kb11, s1);
      }
      // --- masked online softmax update (per-row lane-aligned with C layout)
      unsigned wword = (unsigned)(cbase >> 5);
      float corr[8];
#pragma unroll
      for (int j = 0; j < 8; ++j) {
        int q = q0 + j + 8 * half;
        unsigned mw = mb[((size_t)b * NQ + q) * 64 + wword];
        bool b0 = (mw >> lan) & 1u;
        bool b1 = (mw >> (16 + lan)) & 1u;
        float x0 = b0 ? s0[j] * NORMF : -__builtin_inff();
        float x1 = b1 ? s1[j] * NORMF : -__builtin_inff();
        float t = rmax16(fmaxf(x0, x1));
        float mn = fmaxf(m[j], t);
        float cfac = (mn == -__builtin_inff()) ? 1.0f : __expf(m[j] - mn);
        float p0 = b0 ? __expf(x0 - mn) : 0.0f;
        float p1 = b1 ? __expf(x1 - mn) : 0.0f;
        l[j] = l[j] * cfac + rsum16(p0 + p1);
        m[j] = mn;
        corr[j] = cfac;
        int row = j + 8 * half;
        myp[row * 32 + lan]      = (__bf16)p0;
        myp[row * 32 + 16 + lan] = (__bf16)p1;
      }
#pragma unroll
      for (int vt = 0; vt < 4; ++vt)
#pragma unroll
        for (int j = 0; j < 8; ++j) O[vt][j] *= corr[j];

      // DS stores above must land before re-reading P as an A-fragment
      asm volatile("s_wait_dscnt 0" ::: "memory");

      v16bf pa;
      {
        const __bf16* ap = myp + lan * 32 + half * 16;
        v8bf lo = *(const v8bf*)(ap);
        v8bf hi = *(const v8bf*)(ap + 8);
#pragma unroll
        for (int i = 0; i < 8; ++i) { pa[i] = lo[i]; pa[i + 8] = hi[i]; }
      }
      asm volatile("s_wait_dscnt 0" ::: "memory");

      // --- O += P(16x32) * V(32x64), V pre-transposed so B loads are contiguous
      const __bf16* vrow = Vb_base + (size_t)lan * CTOT + cbase + half * 16;
#pragma unroll
      for (int vt = 0; vt < 4; ++vt) {
        v16bf vb = *(const v16bf*)(vrow + (size_t)vt * 16 * CTOT);
        O[vt] = wmma_bf16(pa, vb, O[vt]);
      }
    }
  }

  // normalize (den>0 guard like reference) and write heads[b][q][h*64+v] bf16
#pragma unroll
  for (int j = 0; j < 8; ++j) {
    float inv = (l[j] > 0.0f) ? 1.0f / l[j] : 0.0f;
#pragma unroll
    for (int vt = 0; vt < 4; ++vt) O[vt][j] *= inv;
  }
#pragma unroll
  for (int vt = 0; vt < 4; ++vt)
#pragma unroll
    for (int j = 0; j < 8; ++j) {
      int row = q0 + j + 8 * half;
      int col = h * 64 + vt * 16 + lan;
      heads[((size_t)b * NQ + row) * (HH * VDIM) + col] = (__bf16)O[vt][j];
    }
}

// ---------------------------------------------------------------------------
// 7) Output GEMM: out(8192x512) = heads(8192x512 bf16) * W_out^T, f32 out
// ---------------------------------------------------------------------------
__global__ __launch_bounds__(256) void out_kernel(
    const __bf16* __restrict__ heads, const __bf16* __restrict__ WoT,
    float* __restrict__ out) {
  int lane = threadIdx.x & 31;
  int half = lane >> 4, lan = lane & 15;
  int w = blockIdx.x * 8 + (threadIdx.x >> 5);
  int ntile = w & 31;
  int mtile = w >> 5;
  int mrow = mtile * 16;

  const __bf16* arow = heads + (size_t)(mrow + lan) * (HH * VDIM);
  const __bf16* brow = WoT + (size_t)(ntile * 16 + lan) * (HH * KD);

  v8f c = zero8();
#pragma unroll 4
  for (int kk = 0; kk < (HH * KD) / 32; ++kk) {
    v16bf a = load_a_frag(arow, half, kk * 32);
    v16bf b = *(const v16bf*)(brow + kk * 32 + half * 16);
    c = wmma_bf16(a, b, c);
  }
#pragma unroll
  for (int j = 0; j < 8; ++j) {
    int row = mrow + j + 8 * half;
    out[(size_t)row * EE + ntile * 16 + lan] = c[j];
  }
}

// ---------------------------------------------------------------------------
// Launch
// ---------------------------------------------------------------------------
extern "C" void kernel_launch(void* const* d_in, const int* in_sizes, int n_in,
                              void* d_out, int out_size, void* d_ws, size_t ws_size,
                              hipStream_t stream) {
  (void)in_sizes; (void)n_in; (void)out_size;
  const float* q  = (const float*)d_in[0];
  const int* att  = (const int*)d_in[1];
  const int* grp  = (const int*)d_in[2];
  const int* sp   = (const int*)d_in[3];
  W12 w12;
  for (int i = 0; i < 12; ++i) w12.p[i] = (const float*)d_in[4 + i];
  const float* wout = (const float*)d_in[16];
  float* out = (float*)d_out;

  char* ws = (char*)d_ws;
  size_t off = 0;
  auto take = [&](size_t bytes) -> char* {
    char* p = ws + off;
    off += (bytes + 255) & ~(size_t)255;
    return p;
  };
  __bf16* qb     = (__bf16*)take((size_t)MROWS * DD * 2);          //  8.4 MB
  __bf16* Wt     = (__bf16*)take((size_t)12 * HH * KD * DD * 2);   //  6.3 MB
  __bf16* WoT    = (__bf16*)take((size_t)EE * HH * KD * 2);        //  0.5 MB
  __bf16* Qb     = (__bf16*)take(4 * QKSZ * 2);                    // 33.6 MB
  __bf16* Kb     = (__bf16*)take(4 * QKSZ * 2);                    // 33.6 MB
  __bf16* Vt     = (__bf16*)take((size_t)HH * BB * VDIM * CTOT * 2);// 33.6 MB
  unsigned* mbit = (unsigned*)take((size_t)BB * NQ * 64 * 4);      //  2.1 MB
  __bf16* heads  = (__bf16*)take((size_t)MROWS * HH * VDIM * 2);   //  8.4 MB
  if (off > ws_size) return;  // workspace too small; nothing safe to do

  qconv_kernel<<<(MROWS * DD) / 256, 256, 0, stream>>>(q, qb);
  wconv_kernel<<<(12 * 8 * 64 * 512) / 256, 256, 0, stream>>>(w12, Wt);
  woutconv_kernel<<<(EE * HH * KD) / 256, 256, 0, stream>>>(wout, WoT);
  maskpack_kernel<<<(BB * NQ * 64) / 256, 256, 0, stream>>>(att, grp, sp, mbit);

  // 12 GEMMs: 12 * 8 heads * 512 mtiles * 4 ntiles = 196608 waves, 8 waves/block
  proj_kernel<<<196608 / 8, 256, 0, stream>>>(qb, Wt, Qb, Kb, Vt);

  // fused attention: 8*16*32 = 4096 waves, 4 waves/block
  attn_kernel<<<4096 / 4, 128, 0, stream>>>(Qb, Kb, Vt, mbit, heads);

  // output GEMM: 512 mtiles * 32 ntiles = 16384 waves, 8 waves/block
  out_kernel<<<16384 / 8, 256, 0, stream>>>(heads, WoT, out);
}
